// LTCNController_45234595561880
// MI455X (gfx1250) — matrix operand order repaired and checked by reading, not compiled, not run on hardware
//
#include <hip/hip_runtime.h>

// LTC Neural Controller, MI455X (gfx1250), wave32 WMMA f16 16x16x32 / f32 accum.
// Weights are pre-packed once into WMMA-fragment-ready f16 buffers (k_prep*),
// then每 GEMM block DMAs its B-slab into LDS with TENSOR_LOAD_TO_LDS (TDM).
//
// k_prepB/k_prepConv: f32 weights -> fragment-ready f16 global buffers
// k_conv1: frames -> implicit GEMM (K=27 pad 32)        -> out1 f16 NCHW
// k_conv2: spatial-major-K implicit GEMM (9 ksteps)     -> out2 f16 NCHW
// k_conv3: spatial-major-K implicit GEMM (18 ksteps)    -> out3 f16 (img,co,pix)
// k_pool : mean over 64 pixels (commutes w/ encoder)    -> pooledF (A-frag layout)
// k_enc  : pooled @ enc_w^T + enc_b                     -> xencF (A-frag, scan order)
// k_xwi  : xenc @ ltc_wi^T + ltc_b                      -> preX f32
// k_scan : WMMA scan, Wr B-fragments resident in VGPRs  -> hseqF (A-frag) + finalH
// k_dec1 : relu(hseq @ dec_w1^T + b1)                   -> y1F (A-frag)
// k_dec2 : relu(y1 @ dec_w2^T + b2)                     -> y2 f16 linear
// k_dec3 : y2 @ dec_w3^T + b3 (N=6, VALU)               -> controls f32

typedef __attribute__((ext_vector_type(16))) _Float16 v16h;
typedef __attribute__((ext_vector_type(8)))  _Float16 v8h;
typedef __attribute__((ext_vector_type(8)))  float    v8f;
typedef unsigned int u32x4 __attribute__((ext_vector_type(4)));
typedef int          i32x4 __attribute__((ext_vector_type(4)));
typedef int          i32x8 __attribute__((ext_vector_type(8)));

#define BB 8
#define TT 256
#define HIDN 256
#define OUTC 6
#define BT (BB * TT)

__device__ __forceinline__ v8f wmma16(v16h a, v16h b, v8f c) {
  return __builtin_amdgcn_wmma_f32_16x16x32_f16(false, a, false, b, (short)0, c,
                                                false, false);
}
// A-fragment: element e of lane-half h covers K = (e&7) + ((e&8)<<1) + (h<<3)
__device__ __forceinline__ int kfragA(int e, int half) {
  return (e & 7) + ((e & 8) << 1) + (half << 3);
}

// ---------- WMMA-fragment-resident A buffers (global) ----------
template <int KS>
__device__ __forceinline__ void storeAfrag(_Float16* buf, int row, int col, float v) {
  const int mt = row >> 4, rl = row & 15;
  const int ks = col >> 5, k = col & 31;
  const int lane = rl + (((k >> 3) & 1) << 4);
  const int e = (k & 7) + ((k & 16) >> 1);
  buf[((((long)mt * KS + ks) << 5) + lane) * 16 + e] = (_Float16)v;
}
template <int KS>
__device__ __forceinline__ v16h loadAfrag(const _Float16* buf, int mt, int ks, int lane) {
  return *(const v16h*)&buf[((((long)mt * KS + ks) << 5) + lane) << 4];
}
template <int KS>
__device__ __forceinline__ v16h loadBfrag(const _Float16* ws, int nt, int ks, int lane) {
  return *(const v16h*)&ws[(((nt * KS + ks) << 5) + lane) << 4];
}

// ---------- TDM: one-shot DMA of a contiguous block into LDS ----------
#if __has_builtin(__builtin_amdgcn_tensor_load_to_lds) && \
    __has_builtin(__builtin_amdgcn_s_wait_tensorcnt)
#define HAVE_TDM 1
#else
#define HAVE_TDM 0
#endif

__device__ __forceinline__ void ldsFill(_Float16* lds, const _Float16* gsrc, int nbytes) {
#if HAVE_TDM
  if (threadIdx.x == 0) {
    const unsigned long long ga = (unsigned long long)(const void*)gsrc;
    const unsigned units = (unsigned)nbytes >> 3; // data_size = 8 bytes
    u32x4 g0 = {};
    g0[0] = 1u;                                       // count=1, user descriptor
    g0[1] = (unsigned)(size_t)(void*)lds;             // lds_addr (aperture low 32b)
    g0[2] = (unsigned)(ga & 0xffffffffu);             // global_addr[31:0]
    g0[3] = (unsigned)((ga >> 32) & 0x1ffffffu) | (2u << 30); // addr[56:32] | type=2
    i32x8 g1 = {};
    g1[0] = (int)(3u << 16);                          // wg_mask=0, data_size=8B
    g1[1] = (int)((units & 0xffffu) << 16);           // tensor_dim0[15:0]
    g1[2] = (int)((units >> 16) & 0xffffu) | (1 << 16); // tensor_dim0[31:16], dim1=1
    g1[3] = (int)(units << 16);                       // tile_dim0 (<=65535 units)
    g1[5] = (int)units;                               // tensor_dim0_stride[31:0]
    i32x4 g2 = {}, g3 = {};
#if defined(__clang_major__) && (__clang_major__ >= 23)
    i32x8 g4 = {};
    __builtin_amdgcn_tensor_load_to_lds(g0, g1, g2, g3, g4, 0);
#else
    __builtin_amdgcn_tensor_load_to_lds(g0, g1, g2, g3, 0);
#endif
    __builtin_amdgcn_s_wait_tensorcnt(0);
  }
  __syncthreads();
#else
  const int n16 = nbytes >> 4;
  for (int i = threadIdx.x; i < n16; i += 256)
    ((v8h*)lds)[i] = ((const v8h*)gsrc)[i];
  __syncthreads();
#endif
}

// ---------- weight prep: fragment-ready f16 buffers ----------
// identity K mapping (zero pad past Kreal); layout [nt][ks][lane][e]
__global__ __launch_bounds__(256) void k_prepB(const float* __restrict__ W,
                                               _Float16* __restrict__ dst, int Kreal,
                                               int KS, int total) {
  const int fi = blockIdx.x * 256 + threadIdx.x;
  if (fi >= total) return;
  const int e = fi & 15, lane = (fi >> 4) & 31;
  const int ks = (fi >> 9) % KS, nt = (fi >> 9) / KS;
  const int n = nt * 16 + (lane & 15);
  const int k = ks * 32 + (e & 7) + ((e & 8) << 1) + ((lane >> 4) << 3); // kfragB = kfragA(e, half)
  dst[fi] = (_Float16)((k < Kreal) ? W[(long)n * Kreal + k] : 0.f);
}
// conv K permutation: q = sp*CI + ci  ->  source k = ci*9 + sp
__global__ __launch_bounds__(256) void k_prepConv(const float* __restrict__ W,
                                                  _Float16* __restrict__ dst, int ciBits,
                                                  int KS, int total) {
  const int fi = blockIdx.x * 256 + threadIdx.x;
  if (fi >= total) return;
  const int e = fi & 15, lane = (fi >> 4) & 31;
  const int ks = (fi >> 9) % KS, nt = (fi >> 9) / KS;
  const int n = nt * 16 + (lane & 15);
  const int q = ks * 32 + (e & 7) + ((e & 8) << 1) + ((lane >> 4) << 3);
  const int ci = q & ((1 << ciBits) - 1), sp = q >> ciBits;
  dst[fi] = (_Float16)W[(long)n * (9 << ciBits) + ci * 9 + sp];
}

// =============================== conv1 ===============================
// 3->32ch 64x64->32x32 s2 p1. K = 27 (ci*9+ky*3+kx) padded to 32.
__global__ __launch_bounds__(256) void k_conv1(const float* __restrict__ frames,
                                               const _Float16* __restrict__ wF,
                                               const float* __restrict__ bias,
                                               _Float16* __restrict__ out1) {
  __shared__ __align__(32) _Float16 wsF[2 * 512]; // 2 KB
  ldsFill(wsF, wF, 2 * 512 * 2);
  const int lane = threadIdx.x & 31, wv = threadIdx.x >> 5;
  const int half = lane >> 4, l15 = lane & 15;
  const long mBase = ((long)blockIdx.x * 8 + wv) * 16;
  const long m = mBase + l15;
  const int img = (int)(m >> 10), p = (int)(m & 1023);
  const int oy2 = (p >> 5) * 2 - 1, ox2 = (p & 31) * 2 - 1;
  const long fb = (long)img * 3 * 4096;
  v16h a;
#pragma unroll
  for (int e = 0; e < 16; ++e) {
    const int k = kfragA(e, half);
    const int ci = k / 9, r = k - ci * 9, ky = r / 3, kx = r - ky * 3;
    const int y = oy2 + ky, x = ox2 + kx;
    const bool ok = (k < 27) & ((unsigned)y < 64u) & ((unsigned)x < 64u);
    const int off = ok ? (ci * 4096 + y * 64 + x) : 0; // unconditional load
    const float v = frames[fb + off];
    a[e] = ok ? (_Float16)v : (_Float16)0.f;
  }
#pragma unroll
  for (int nt = 0; nt < 2; ++nt) {
    v8f c = {};
    c = wmma16(a, loadBfrag<1>(wsF, nt, 0, lane), c);
    const int co = nt * 16 + l15;
    const float bb = bias[co];
    v8h pk;
#pragma unroll
    for (int v = 0; v < 8; ++v) {
      float r = c[v] + bb;
      pk[v] = (_Float16)(r > 0.f ? r : 0.f);
    }
    const long mm0 = mBase + half * 8; // 8 consecutive pixels in one row
    const int i2 = (int)(mm0 >> 10), pp0 = (int)(mm0 & 1023);
    *(v8h*)&out1[(((long)i2 * 32 + co) * 32 + (pp0 >> 5)) * 32 + (pp0 & 31)] = pk;
  }
}

// =============================== conv2 ===============================
// 32->64ch 32x32->16x16. K spatial-major: q = (ky*3+kx)*32 + ci.
__global__ __launch_bounds__(256) void k_conv2(const _Float16* __restrict__ in,
                                               const _Float16* __restrict__ wF,
                                               const float* __restrict__ bias,
                                               _Float16* __restrict__ out) {
  __shared__ __align__(32) _Float16 wsF[4 * 9 * 512]; // 36 KB
  ldsFill(wsF, wF, 4 * 9 * 512 * 2);
  const int lane = threadIdx.x & 31, wv = threadIdx.x >> 5;
  const int half = lane >> 4, l15 = lane & 15;
  const long mBase = ((long)blockIdx.x * 8 + wv) * 16;
  const long m = mBase + l15;
  const int img = (int)(m >> 8), p = (int)(m & 255);
  const int oy2 = (p >> 4) * 2 - 1, ox2 = (p & 15) * 2 - 1;
  const long ib = (long)img * 32 * 1024;

  v8f acc[4] = {};
#pragma unroll
  for (int ks = 0; ks < 9; ++ks) {
    const int ky = ks / 3, kx = ks - ky * 3; // compile-time after unroll
    const int y = oy2 + ky, x = ox2 + kx;
    const bool ok = ((unsigned)y < 32u) & ((unsigned)x < 32u);
    const int yx = ok ? (y * 32 + x) : 0;
    v16h a;
#pragma unroll
    for (int e = 0; e < 16; ++e) {
      const _Float16 v = in[ib + kfragA(e, half) * 1024 + yx]; // ci = frag pattern
      a[e] = ok ? v : (_Float16)0.f;
    }
#pragma unroll
    for (int nt = 0; nt < 4; ++nt)
      acc[nt] = wmma16(a, loadBfrag<9>(wsF, nt, ks, lane), acc[nt]);
  }
#pragma unroll
  for (int nt = 0; nt < 4; ++nt) {
    const int co = nt * 16 + l15;
    const float bb = bias[co];
    v8h pk;
#pragma unroll
    for (int v = 0; v < 8; ++v) {
      float r = acc[nt][v] + bb;
      pk[v] = (_Float16)(r > 0.f ? r : 0.f);
    }
    const long mm0 = mBase + half * 8;
    const int i2 = (int)(mm0 >> 8), pp0 = (int)(mm0 & 255);
    *(v8h*)&out[(((long)i2 * 64 + co) * 16 + (pp0 >> 4)) * 16 + (pp0 & 15)] = pk;
  }
}

// =============================== conv3 ===============================
// 64->128ch 16x16->8x8. q = (ky*3+kx)*64 + ci; N split by grid.y halves.
__global__ __launch_bounds__(256) void k_conv3(const _Float16* __restrict__ in,
                                               const _Float16* __restrict__ wF,
                                               const float* __restrict__ bias,
                                               _Float16* __restrict__ out) {
  __shared__ __align__(32) _Float16 wsF[4 * 18 * 512]; // 72 KB
  const int nOff = blockIdx.y * 64;
  ldsFill(wsF, wF + (long)blockIdx.y * 4 * 18 * 512, 4 * 18 * 512 * 2);
  const int lane = threadIdx.x & 31, wv = threadIdx.x >> 5;
  const int half = lane >> 4, l15 = lane & 15;
  const long mBase = ((long)blockIdx.x * 8 + wv) * 16;
  const long m = mBase + l15;
  const int img = (int)(m >> 6), p = (int)(m & 63);
  const int oy2 = (p >> 3) * 2 - 1, ox2 = (p & 7) * 2 - 1;
  const long ib = (long)img * 64 * 256;

  v8f acc[4] = {};
#pragma unroll
  for (int ks = 0; ks < 18; ++ks) {
    const int sp = ks >> 1, ciB = (ks & 1) << 5;
    const int ky = sp / 3, kx = sp - ky * 3;
    const int y = oy2 + ky, x = ox2 + kx;
    const bool ok = ((unsigned)y < 16u) & ((unsigned)x < 16u);
    const int yx = ok ? (y * 16 + x) : 0;
    v16h a;
#pragma unroll
    for (int e = 0; e < 16; ++e) {
      const _Float16 v = in[ib + (ciB + kfragA(e, half)) * 256 + yx];
      a[e] = ok ? v : (_Float16)0.f;
    }
#pragma unroll
    for (int nt = 0; nt < 4; ++nt)
      acc[nt] = wmma16(a, loadBfrag<18>(wsF, nt, ks, lane), acc[nt]);
  }
#pragma unroll
  for (int nt = 0; nt < 4; ++nt) {
    const int co = nOff + nt * 16 + l15;
    const float bb = bias[co];
    v8h pk;
#pragma unroll
    for (int v = 0; v < 8; ++v) {
      float r = acc[nt][v] + bb;
      pk[v] = (_Float16)(r > 0.f ? r : 0.f);
    }
    const long mm0 = mBase + half * 8;
    const int i2 = (int)(mm0 >> 6), pp0 = (int)(mm0 & 63);
    *(v8h*)&out[((long)i2 * 128 + co) * 64 + pp0] = pk; // (img, co, pix)
  }
}

// =============================== pool ===============================
__global__ __launch_bounds__(256) void k_pool(const _Float16* __restrict__ in,
                                              _Float16* __restrict__ pooledF) {
  const int idx = blockIdx.x * 256 + threadIdx.x; // BT*128
  const int img = idx >> 7, co = idx & 127;
  const _Float16* src = in + ((long)img * 128 + co) * 64; // contiguous 128B
  float s = 0.f;
#pragma unroll
  for (int t = 0; t < 8; ++t) {
    const v8h v = *(const v8h*)&src[t * 8];
#pragma unroll
    for (int j = 0; j < 8; ++j) s += (float)v[j];
  }
  storeAfrag<4>(pooledF, img, co, s * (1.f / 64.f));
}

// =============================== encoder ===============================
// M=2048 (rows b*T+t), K=128, N=256; output rows remapped to scan order t*8+b.
__global__ __launch_bounds__(256) void k_enc(const _Float16* __restrict__ pooledF,
                                             const _Float16* __restrict__ wF,
                                             const float* __restrict__ bias,
                                             _Float16* __restrict__ xencF) {
  __shared__ __align__(32) _Float16 wsF[4 * 4 * 512]; // 16 KB
  ldsFill(wsF, wF + (long)blockIdx.y * 4 * 4 * 512, 4 * 4 * 512 * 2);
  const int lane = threadIdx.x & 31, wv = threadIdx.x >> 5;
  const int half = lane >> 4, l15 = lane & 15;
  const int mt = blockIdx.x * 8 + wv;
  const long mBase = (long)mt * 16;

  v8f acc[4] = {};
#pragma unroll
  for (int ks = 0; ks < 4; ++ks) {
    const v16h a = loadAfrag<4>(pooledF, mt, ks, lane);
#pragma unroll
    for (int nt = 0; nt < 4; ++nt)
      acc[nt] = wmma16(a, loadBfrag<4>(wsF, nt, ks, lane), acc[nt]);
  }
#pragma unroll
  for (int nt = 0; nt < 4; ++nt) {
    const int n = blockIdx.y * 64 + nt * 16 + l15;
    const float bb = bias[n];
#pragma unroll
    for (int v = 0; v < 8; ++v) {
      const long mm = mBase + v + half * 8;              // mm = b*T + t
      const int row = (int)((mm & 255) * 8 + (mm >> 8)); // t*8 + b
      storeAfrag<8>(xencF, row, n, acc[nt][v] + bb);
    }
  }
}

// =============================== x @ Wi^T + ltc_b ===============================
__global__ __launch_bounds__(256) void k_xwi(const _Float16* __restrict__ xencF,
                                             const _Float16* __restrict__ wF,
                                             const float* __restrict__ ltc_b,
                                             float* __restrict__ preX) {
  __shared__ __align__(32) _Float16 wsF[4 * 8 * 512]; // 32 KB
  ldsFill(wsF, wF + (long)blockIdx.y * 4 * 8 * 512, 4 * 8 * 512 * 2);
  const int lane = threadIdx.x & 31, wv = threadIdx.x >> 5;
  const int half = lane >> 4, l15 = lane & 15;
  const int mt = blockIdx.x * 8 + wv;
  const long mBase = (long)mt * 16;

  v8f acc[4] = {};
#pragma unroll
  for (int ks = 0; ks < 8; ++ks) {
    const v16h a = loadAfrag<8>(xencF, mt, ks, lane);
#pragma unroll
    for (int nt = 0; nt < 4; ++nt)
      acc[nt] = wmma16(a, loadBfrag<8>(wsF, nt, ks, lane), acc[nt]);
  }
#pragma unroll
  for (int nt = 0; nt < 4; ++nt) {
    const int n = blockIdx.y * 64 + nt * 16 + l15;
    const float bb = ltc_b[n];
#pragma unroll
    for (int v = 0; v < 8; ++v)
      preX[(mBase + v + half * 8) * 256 + n] = acc[nt][v] + bb;
  }
}

// =============================== LTC scan (WMMA) ===============================
// 1 workgroup, 8 waves. Wave w owns columns [w*32, w*32+32): its Wr slab lives in
// VGPRs as 16 pre-packed B fragments for all 256 steps (loaded straight from the
// fragment-ready buffer as b128s). h kept in LDS in A-fragment order (16x256,
// rows 8..15 zero) -> 16 WMMAs per wave per step.
__global__ __launch_bounds__(256) void k_scan(const _Float16* __restrict__ wrF,
                                              const float* __restrict__ h0,
                                              const float* __restrict__ preX,
                                              const float* __restrict__ tau,
                                              const float* __restrict__ aVec,
                                              _Float16* __restrict__ hseqF,
                                              float* __restrict__ finalH) {
  __shared__ __align__(32) _Float16 hA[8 * 512]; // A-frag h, 8 KB
  __shared__ float hF[BB * HIDN];                // f32 h, 8 KB
  const int tid = threadIdx.x;
  const int lane = tid & 31, wv = tid >> 5;
  const int half = lane >> 4, l15 = lane & 15;

  for (int i = tid; i < 8 * 512; i += 256) hA[i] = (_Float16)0.f;
  __syncthreads();
  for (int i = tid; i < BB * HIDN; i += 256) {
    hF[i] = h0[i];
    const int b = i >> 8, j = i & 255;
    const int ks = j >> 5, k = j & 31;
    const int lt = b + (((k >> 3) & 1) << 4);
    const int ee = (k & 7) + ((k & 16) >> 1);
    hA[(((ks << 5) + lt) << 4) + ee] = (_Float16)h0[i];
  }

  v16h Bf[2][8];
#pragma unroll
  for (int nt = 0; nt < 2; ++nt)
#pragma unroll
    for (int ks = 0; ks < 8; ++ks)
      Bf[nt][ks] = *(const v16h*)&wrF[(((((wv * 2 + nt) * 8 + ks)) << 5) + lane) << 4];

  const int col0 = wv * 32 + l15, col1 = col0 + 16;
  const float ti0 = 1.f / tau[col0], ti1 = 1.f / tau[col1];
  const float av0 = aVec[col0], av1 = aVec[col1];
  __syncthreads();

  for (int s = 0; s < TT; ++s) {
    v8f acc0 = {}, acc1 = {};
#pragma unroll
    for (int ks = 0; ks < 8; ++ks) {
      const v16h a = *(const v16h*)&hA[((ks << 5) + lane) << 4];
      acc0 = wmma16(a, Bf[0][ks], acc0);
      acc1 = wmma16(a, Bf[1][ks], acc1);
    }
    __syncthreads(); // all hA reads done before updates
    if (half == 0) { // lanes 0..15 hold batches 0..7 for both n-tiles
#pragma unroll
      for (int nt = 0; nt < 2; ++nt) {
        const int colv = nt ? col1 : col0;
        const float ti = nt ? ti1 : ti0, av = nt ? av1 : av0;
        const int ks = colv >> 5, k = colv & 31;
        const int hb = (((k >> 3) & 1) << 4);
        const int ee = (k & 7) + ((k & 16) >> 1);
#pragma unroll
        for (int b = 0; b < 8; ++b) {
          const float d = nt ? acc1[b] : acc0[b];
          const float f = tanhf(d + preX[((long)s * 8 + b) * 256 + colv]);
          const float h = hF[b * 256 + colv];
          const float hn = h + 0.1f * (-(ti + f) * h + f * av);
          hF[b * 256 + colv] = hn;
          hA[(((ks << 5) + b + hb) << 4) + ee] = (_Float16)hn;
          storeAfrag<8>(hseqF, s * 8 + b, colv, hn);
        }
      }
    }
    __syncthreads();
  }
  for (int i = tid; i < BB * HIDN; i += 256) finalH[i] = hF[i];
}

// =============================== decoder 1 ===============================
__global__ __launch_bounds__(256) void k_dec1(const _Float16* __restrict__ hseqF,
                                              const _Float16* __restrict__ wF,
                                              const float* __restrict__ bias,
                                              _Float16* __restrict__ y1F) {
  __shared__ __align__(32) _Float16 wsF[8 * 8 * 512]; // 64 KB
  ldsFill(wsF, wF, 8 * 8 * 512 * 2);
  const int lane = threadIdx.x & 31, wv = threadIdx.x >> 5;
  const int half = lane >> 4, l15 = lane & 15;
  const int mt = blockIdx.x * 8 + wv;
  const long mBase = (long)mt * 16;

  v8f acc[8] = {};
#pragma unroll
  for (int ks = 0; ks < 8; ++ks) {
    const v16h a = loadAfrag<8>(hseqF, mt, ks, lane);
#pragma unroll
    for (int nt = 0; nt < 8; ++nt)
      acc[nt] = wmma16(a, loadBfrag<8>(wsF, nt, ks, lane), acc[nt]);
  }
#pragma unroll
  for (int nt = 0; nt < 8; ++nt) {
    const int n = nt * 16 + l15;
    const float bb = bias[n];
#pragma unroll
    for (int v = 0; v < 8; ++v) {
      float r = acc[nt][v] + bb;
      storeAfrag<4>(y1F, (int)(mBase + v + half * 8), n, r > 0.f ? r : 0.f);
    }
  }
}

// =============================== decoder 2 ===============================
__global__ __launch_bounds__(256) void k_dec2(const _Float16* __restrict__ y1F,
                                              const _Float16* __restrict__ wF,
                                              const float* __restrict__ bias,
                                              _Float16* __restrict__ y2) {
  __shared__ __align__(32) _Float16 wsF[4 * 4 * 512]; // 16 KB
  ldsFill(wsF, wF, 4 * 4 * 512 * 2);
  const int lane = threadIdx.x & 31, wv = threadIdx.x >> 5;
  const int half = lane >> 4, l15 = lane & 15;
  const int mt = blockIdx.x * 8 + wv;
  const long mBase = (long)mt * 16;

  v8f acc[4] = {};
#pragma unroll
  for (int ks = 0; ks < 4; ++ks) {
    const v16h a = loadAfrag<4>(y1F, mt, ks, lane);
#pragma unroll
    for (int nt = 0; nt < 4; ++nt)
      acc[nt] = wmma16(a, loadBfrag<4>(wsF, nt, ks, lane), acc[nt]);
  }
#pragma unroll
  for (int nt = 0; nt < 4; ++nt) {
    const int n = nt * 16 + l15;
    const float bb = bias[n];
#pragma unroll
    for (int v = 0; v < 8; ++v) {
      float r = acc[nt][v] + bb;
      y2[(mBase + v + half * 8) * 64 + n] = (_Float16)(r > 0.f ? r : 0.f);
    }
  }
}

// =============================== decoder 3 ===============================
__global__ __launch_bounds__(256) void k_dec3(const _Float16* __restrict__ y2,
                                              const float* __restrict__ w,
                                              const float* __restrict__ bias,
                                              float* __restrict__ controls) {
  const int idx = blockIdx.x * 256 + threadIdx.x; // exactly BT*6
  const int m = idx / OUTC, o = idx - m * OUTC;
  const int s = m >> 3, b = m & 7;
  const _Float16* row = y2 + (long)m * 64;
  float acc = bias[o];
#pragma unroll
  for (int t = 0; t < 8; ++t) {
    const v8h v = *(const v8h*)&row[t * 8];
#pragma unroll
    for (int j = 0; j < 8; ++j) acc += (float)v[j] * w[o * 64 + t * 8 + j];
  }
  controls[((long)b * TT + s) * OUTC + o] = acc;
}

// =============================== launch ===============================
extern "C" void kernel_launch(void* const* d_in, const int* in_sizes, int n_in,
                              void* d_out, int out_size, void* d_ws, size_t ws_size,
                              hipStream_t stream) {
  (void)in_sizes; (void)n_in; (void)out_size; (void)ws_size;
  const float* frames = (const float*)d_in[0];
  const float* h0     = (const float*)d_in[1];
  const float* c1w = (const float*)d_in[2];  const float* c1b = (const float*)d_in[3];
  const float* c2w = (const float*)d_in[4];  const float* c2b = (const float*)d_in[5];
  const float* c3w = (const float*)d_in[6];  const float* c3b = (const float*)d_in[7];
  const float* encw = (const float*)d_in[8]; const float* encb = (const float*)d_in[9];
  const float* wr = (const float*)d_in[10];  const float* wi = (const float*)d_in[11];
  const float* lb = (const float*)d_in[12];  const float* ltau = (const float*)d_in[13];
  const float* la = (const float*)d_in[14];
  const float* dw1 = (const float*)d_in[15]; const float* db1 = (const float*)d_in[16];
  const float* dw2 = (const float*)d_in[17]; const float* db2 = (const float*)d_in[18];
  const float* dw3 = (const float*)d_in[19]; const float* db3 = (const float*)d_in[20];

  char* ws = (char*)d_ws;
  _Float16* out1    = (_Float16*)ws; ws += (size_t)BT * 32 * 32 * 32 * 2; // 128 MB
  _Float16* out2    = (_Float16*)ws; ws += (size_t)BT * 64 * 16 * 16 * 2; //  64 MB
  _Float16* out3    = (_Float16*)ws; ws += (size_t)BT * 128 * 64 * 2;     //  32 MB
  _Float16* pooledF = (_Float16*)ws; ws += (size_t)BT * 128 * 2;
  _Float16* xencF   = (_Float16*)ws; ws += (size_t)BT * 256 * 2;
  float*    preX    = (float*)ws;    ws += (size_t)BT * 256 * 4;
  _Float16* hseqF   = (_Float16*)ws; ws += (size_t)BT * 256 * 2;
  _Float16* y1F     = (_Float16*)ws; ws += (size_t)BT * 128 * 2;
  _Float16* y2      = (_Float16*)ws; ws += (size_t)BT * 64 * 2;
  // fragment-ready weight buffers
  _Float16* pc1  = (_Float16*)ws; ws += 1024 * 2;
  _Float16* pc2  = (_Float16*)ws; ws += 18432 * 2;
  _Float16* pc3  = (_Float16*)ws; ws += 73728 * 2;
  _Float16* pEnc = (_Float16*)ws; ws += 32768 * 2;
  _Float16* pXwi = (_Float16*)ws; ws += 65536 * 2;
  _Float16* pWr  = (_Float16*)ws; ws += 65536 * 2;
  _Float16* pD1  = (_Float16*)ws; ws += 32768 * 2;
  _Float16* pD2  = (_Float16*)ws; ws += 8192 * 2;

  float* controls = (float*)d_out;                  // (B,T,6)
  float* finalH   = (float*)d_out + BB * TT * OUTC; // (B,HID)

  // ---- weight prep (one-shot, ~580 KB total) ----
  k_prepB   <<<dim3(4),   dim3(256), 0, stream>>>(c1w, pc1, 27, 1, 1024);
  k_prepConv<<<dim3(72),  dim3(256), 0, stream>>>(c2w, pc2, 5, 9, 18432);
  k_prepConv<<<dim3(288), dim3(256), 0, stream>>>(c3w, pc3, 6, 18, 73728);
  k_prepB   <<<dim3(128), dim3(256), 0, stream>>>(encw, pEnc, 128, 4, 32768);
  k_prepB   <<<dim3(256), dim3(256), 0, stream>>>(wi, pXwi, 256, 8, 65536);
  k_prepB   <<<dim3(256), dim3(256), 0, stream>>>(wr, pWr, 256, 8, 65536);
  k_prepB   <<<dim3(128), dim3(256), 0, stream>>>(dw1, pD1, 256, 8, 32768);
  k_prepB   <<<dim3(32),  dim3(256), 0, stream>>>(dw2, pD2, 128, 4, 8192);

  // ---- pipeline ----
  k_conv1<<<dim3(16384),   dim3(256), 0, stream>>>(frames, pc1, c1b, out1);
  k_conv2<<<dim3(4096),    dim3(256), 0, stream>>>(out1, pc2, c2b, out2);
  k_conv3<<<dim3(1024, 2), dim3(256), 0, stream>>>(out2, pc3, c3b, out3);
  k_pool <<<dim3(1024),    dim3(256), 0, stream>>>(out3, pooledF);
  k_enc  <<<dim3(16, 4),   dim3(256), 0, stream>>>(pooledF, pEnc, encb, xencF);
  k_xwi  <<<dim3(16, 4),   dim3(256), 0, stream>>>(xencF, pXwi, lb, preX);
  k_scan <<<dim3(1),       dim3(256), 0, stream>>>(pWr, h0, preX, ltau, la, hseqF, finalH);
  k_dec1 <<<dim3(16),      dim3(256), 0, stream>>>(hseqF, pD1, db1, y1F);
  k_dec2 <<<dim3(16),      dim3(256), 0, stream>>>(y1F, pD2, db2, y2);
  k_dec3 <<<dim3(48),      dim3(256), 0, stream>>>(y2, dw3, db3, controls);
}
